// cls_model_37958920962407
// MI455X (gfx1250) — matrix operand order repaired and checked by reading, not compiled, not run on hardware
//
#include <hip/hip_runtime.h>
#include <hip/hip_bf16.h>

// ---------------------------------------------------------------------------
// Problem constants (from the reference): B=16, N=2048, FIN=6, R=64, S=64,
// NCLS=6, KNN=40.  d_out = [16*6 logits][9 regs] = 105 floats.
// ---------------------------------------------------------------------------
#define BB   16
#define NN_  2048
#define F0   6
#define RR   64
#define SS   64
#define NCLS 6
#define KNN  40

typedef __attribute__((ext_vector_type(16))) __bf16 v16bf;
typedef __attribute__((ext_vector_type(8)))  __bf16 v8bf;
typedef __attribute__((ext_vector_type(4)))  __bf16 v4bf;
typedef __attribute__((ext_vector_type(8)))  float  v8f;
typedef __attribute__((ext_vector_type(4)))  float  v4f;

union Frag16 { v16bf v; v8bf h[2]; };

// ---------------------------------------------------------------------------
// Batched WMMA GEMM:  C[b] = alpha * op(A[b]) x op(B[b]) (+ beta*Csrc) (+bias)(relu)
//   TRANSA=0: A(m,k) = A[m*lda+k]   TRANSA=1: A(m,k) = A[k*lda+m]
//   TRANSB=0: B(k,n) = B[k*ldb+n]   TRANSB=1: B(k,n) = B[n*ldb+k]
// Tile 64x128x32, 128 threads (4 waves); wave w owns a 16-row stripe with 8
// accumulators (A fragment reused 8x).  f32 accumulate via
// v_wmma_f32_16x16x32_bf16.  LDS is double-buffered: one barrier per K-step,
// next tile stages while current tile computes.
//
// ABF16 (the [B,N,N] bf16 Laplacian stream): A fragments load straight from
// global as aligned 128-bit chunks (rows are wave-private; LDS staging would
// add traffic without reuse), plus a prefetch on the next K chunk.
// Full tiles (all hot GEMMs) take a guard-free vectorized staging path:
// v4f global loads -> packed bf16 converts -> v4bf (b64) LDS stores, with a
// 4x4 register transpose where the LDS layout is transposed vs. memory.
// ---------------------------------------------------------------------------
#define TM 64
#define TN 128
#define TK 32
#define NACC (TN / 16)   // 8 accumulators per wave
#define PITCH 40         // bf16 elements per LDS tile row (80 B, 16-B aligned)

template<int TRANSA, int TRANSB, bool ABF16>
__global__ void gemm_wmma(const void* __restrict__ Aptr,
                          const float* __restrict__ Bp,
                          float* __restrict__ Cp,
                          const float* __restrict__ Csrc,
                          const float* __restrict__ bias,
                          int M, int N, int K,
                          int lda, int ldb, int ldc, int ldcs,
                          long long sA, long long sB, long long sC, long long sCs,
                          float alpha, float beta, int relu)
{
    __shared__ __bf16 As[2][ABF16 ? 1 : TM][PITCH] __attribute__((aligned(16)));
    __shared__ __bf16 Bs[2][TN][PITCH]             __attribute__((aligned(16)));  // n-major

    const int tid  = threadIdx.x;
    const int wave = tid >> 5;
    const int lane = tid & 31;
    const int lo   = lane & 15;
    const int hi   = lane >> 4;

    const int mBase = blockIdx.y * TM;
    const int nBase = blockIdx.x * TN;
    const int rBase = wave * 16;            // wave's 16-row stripe in the tile

    const __bf16* Ab = (const __bf16*)Aptr;
    const float*  Af = (const float*)Aptr;
    const long long aOff  = (long long)blockIdx.z * sA;
    const long long bOff  = (long long)blockIdx.z * sB;
    const long long cOff  = (long long)blockIdx.z * sC;
    const long long csOff = (long long)blockIdx.z * sCs;

    // ABF16 path: per-lane row pointer (row = mBase+rBase+lo, same for both
    // lane halves; hi selects the K sub-chunks).  All ABF16 calls have M,K
    // multiples of the tile so no bounds handling is needed on A.
    const __bf16* Arow = ABF16
        ? (Ab + aOff + (long long)(mBase + rBase + lo) * lda)
        : nullptr;

    // ---- staging helpers -------------------------------------------------
    auto stageA = [&](int bsel, int kk0) {
        if (ABF16) return;
        bool full = (mBase + TM <= M) && (kk0 + TK <= K) && ((lda & 3) == 0);
        if (full) {
            if (TRANSA == 0) {
                // contiguous along k: v4f load -> v4bf store
                #pragma unroll
                for (int i = 0; i < (TM * TK) / (4 * 128); ++i) {   // 4 iters
                    int idx = tid + i * 128;
                    int m = idx >> 3, kp = (idx & 7) * 4;
                    v4f r = *(const v4f*)(Af + aOff + (long long)(mBase + m) * lda + kk0 + kp);
                    v4bf c = { (__bf16)r[0], (__bf16)r[1], (__bf16)r[2], (__bf16)r[3] };
                    *(v4bf*)&As[bsel][m][kp] = c;
                }
            } else {
                // contiguous along m: 4x4 register transpose per thread
                int bm = (tid & 15) * 4, bk = (tid >> 4) * 4;
                const float* base = Af + aOff + (long long)(kk0 + bk) * lda + mBase + bm;
                v4f r0 = *(const v4f*)(base);
                v4f r1 = *(const v4f*)(base + lda);
                v4f r2 = *(const v4f*)(base + 2 * (long long)lda);
                v4f r3 = *(const v4f*)(base + 3 * (long long)lda);
                #pragma unroll
                for (int j = 0; j < 4; ++j) {
                    v4bf c = { (__bf16)r0[j], (__bf16)r1[j], (__bf16)r2[j], (__bf16)r3[j] };
                    *(v4bf*)&As[bsel][bm + j][bk] = c;
                }
            }
        } else {
            for (int i = tid; i < TM * TK; i += 128) {
                int r = i / TK, c = i % TK;
                int gm = mBase + r, gk = kk0 + c;
                float v = 0.f;
                if (gm < M && gk < K) {
                    long long idx = TRANSA ? (aOff + (long long)gk * lda + gm)
                                           : (aOff + (long long)gm * lda + gk);
                    v = Af[idx];
                }
                As[bsel][r][c] = (__bf16)v;
            }
        }
    };

    auto stageB = [&](int bsel, int kk0) {
        bool full = (nBase + TN <= N) && (kk0 + TK <= K) && ((ldb & 3) == 0);
        if (full) {
            if (TRANSB == 0) {
                // memory k-major, LDS n-major: 4x4 register transpose
                #pragma unroll
                for (int i = 0; i < (TN * TK) / (16 * 128); ++i) {  // 2 iters
                    int idx = tid + i * 128;
                    int bn = (idx & 31) * 4, bk = (idx >> 5) * 4;
                    const float* base = Bp + bOff + (long long)(kk0 + bk) * ldb + nBase + bn;
                    v4f r0 = *(const v4f*)(base);
                    v4f r1 = *(const v4f*)(base + ldb);
                    v4f r2 = *(const v4f*)(base + 2 * (long long)ldb);
                    v4f r3 = *(const v4f*)(base + 3 * (long long)ldb);
                    #pragma unroll
                    for (int j = 0; j < 4; ++j) {
                        v4bf c = { (__bf16)r0[j], (__bf16)r1[j], (__bf16)r2[j], (__bf16)r3[j] };
                        *(v4bf*)&Bs[bsel][bn + j][bk] = c;
                    }
                }
            } else {
                // memory k-contiguous already: straight v4f -> v4bf
                #pragma unroll
                for (int i = 0; i < (TN * TK) / (4 * 128); ++i) {   // 8 iters
                    int idx = tid + i * 128;
                    int n = idx >> 3, kp = (idx & 7) * 4;
                    v4f r = *(const v4f*)(Bp + bOff + (long long)(nBase + n) * ldb + kk0 + kp);
                    v4bf c = { (__bf16)r[0], (__bf16)r[1], (__bf16)r[2], (__bf16)r[3] };
                    *(v4bf*)&Bs[bsel][n][kp] = c;
                }
            }
        } else {
            for (int i = tid; i < TK * TN; i += 128) {
                int n, kk;
                if (TRANSB) { n = i / TK; kk = i % TK; }
                else        { kk = i / TN; n = i % TN; }
                int gk = kk0 + kk, gn = nBase + n;
                float v = 0.f;
                if (gk < K && gn < N) {
                    long long idx = TRANSB ? (bOff + (long long)gn * ldb + gk)
                                           : (bOff + (long long)gk * ldb + gn);
                    v = Bp[idx];
                }
                Bs[bsel][n][kk] = (__bf16)v;
            }
        }
    };

    v8f acc[NACC] = {};

    stageA(0, 0);
    stageB(0, 0);
    __syncthreads();

    int buf = 0;
    for (int k0 = 0; k0 < K; k0 += TK) {
        int kn = k0 + TK;
        if (kn < K) {            // stage next tile into the other buffer
            stageA(buf ^ 1, kn);
            stageB(buf ^ 1, kn);
        }

        // ---- A fragment: two aligned 128-bit loads ----
        Frag16 a;
        if (ABF16) {
            a.h[0] = *(const v8bf*)(Arow + k0 + hi * 8);
            a.h[1] = *(const v8bf*)(Arow + k0 + 16 + hi * 8);
            if (kn < K)
                __builtin_prefetch(Arow + kn, 0, 0);        // global_prefetch
        } else {
            const __bf16* ar = &As[buf][rBase + lo][0];
            a.h[0] = *(const v8bf*)(ar + hi * 8);
            a.h[1] = *(const v8bf*)(ar + 16 + hi * 8);
        }

        // ---- 8 x (B fragment: two ds_load_b128 + WMMA) ----
        #pragma unroll
        for (int t = 0; t < NACC; ++t) {
            Frag16 b;
            const __bf16* br = &Bs[buf][t * 16 + lo][0];
            b.h[0] = *(const v8bf*)(br + hi * 8);
            b.h[1] = *(const v8bf*)(br + 16 + hi * 8);
            acc[t] = __builtin_amdgcn_wmma_f32_16x16x32_bf16(
                         false, a.v, false, b.v, (short)0, acc[t], false, false);
        }
        __syncthreads();
        buf ^= 1;
    }

    // ---- epilogue ----
    #pragma unroll
    for (int t = 0; t < NACC; ++t) {
        #pragma unroll
        for (int v = 0; v < 8; ++v) {
            int gm = mBase + rBase + v + hi * 8;
            int gn = nBase + t * 16 + lo;
            if (gm < M && gn < N) {
                float r = alpha * acc[t][v];
                if (Csrc) r += beta * Csrc[csOff + (long long)gm * ldcs + gn];
                if (bias) r += bias[gn];
                if (relu) r = fmaxf(r, 0.f);
                Cp[cOff + (long long)gm * ldc + gn] = r;
            }
        }
    }
}

// ---------------------------------------------------------------------------
// xx[i] = sum_f X[i,f]^2  (i over B*rows)
// ---------------------------------------------------------------------------
__global__ void sqnorm_rows(const float* __restrict__ X, float* __restrict__ xx,
                            long long nrows, int F)
{
    long long i = (long long)blockIdx.x * blockDim.x + threadIdx.x;
    if (i >= nrows) return;
    const float* p = X + i * F;
    float s = 0.f;
    for (int f = 0; f < F; ++f) s += p[f] * p[f];
    xx[i] = s;
}

// ---------------------------------------------------------------------------
// Per row: a = exp(-(xx_n + xx_m - 2*G)); top-k threshold via binary search
// on value (a in (0,1]); zero below threshold; row-sum -> dsum.
// Overwrites G with the sparsified affinity A.
// ---------------------------------------------------------------------------
__global__ void affinity_topk(float* __restrict__ G, const float* __restrict__ xx,
                              float* __restrict__ dsum, int k)
{
    __shared__ float a[NN_];
    __shared__ float redf[256];
    __shared__ int   redi[256];

    const int row = blockIdx.x;          // b*N + n
    const int b   = row / NN_;
    const int tid = threadIdx.x;
    float* g = G + (long long)row * NN_;
    const float* xb = xx + (long long)b * NN_;
    const float xn  = xx[row];

    float lmax = 0.f;
    for (int m = tid; m < NN_; m += 256) {
        float d = xn + xb[m] - 2.f * g[m];
        float v = __expf(-d);
        a[m] = v;
        lmax = fmaxf(lmax, v);
    }
    redf[tid] = lmax; __syncthreads();
    for (int s = 128; s; s >>= 1) {
        if (tid < s) redf[tid] = fmaxf(redf[tid], redf[tid + s]);
        __syncthreads();
    }
    float lo = 0.f, hi = redf[0];
    __syncthreads();

    for (int it = 0; it < 30; ++it) {
        float mid = 0.5f * (lo + hi);
        int cnt = 0;
        for (int m = tid; m < NN_; m += 256) cnt += (a[m] >= mid) ? 1 : 0;
        redi[tid] = cnt; __syncthreads();
        for (int s = 128; s; s >>= 1) {
            if (tid < s) redi[tid] += redi[tid + s];
            __syncthreads();
        }
        int total = redi[0]; __syncthreads();
        if (total >= k) lo = mid; else hi = mid;
    }

    float ls = 0.f;
    for (int m = tid; m < NN_; m += 256) {
        float v = (a[m] >= lo) ? a[m] : 0.f;
        g[m] = v;
        ls += v;
    }
    redf[tid] = ls; __syncthreads();
    for (int s = 128; s; s >>= 1) {
        if (tid < s) redf[tid] += redf[tid + s];
        __syncthreads();
    }
    if (tid == 0) dsum[row] = redf[0];
}

// ---------------------------------------------------------------------------
// L = I - D^{-1/2} A D^{-1/2}, stored bf16 (halves the HBM stream on the
// dominant [B,N,N] operand).
// ---------------------------------------------------------------------------
__global__ void build_lap_bf16(const float* __restrict__ A,
                               const float* __restrict__ dsum,
                               __bf16* __restrict__ L)
{
    long long idx = (long long)blockIdx.x * blockDim.x + threadIdx.x;
    long long total = (long long)BB * NN_ * NN_;
    if (idx >= total) return;
    long long b = idx / ((long long)NN_ * NN_);
    long long r = idx % ((long long)NN_ * NN_);
    int n = (int)(r / NN_), m = (int)(r % NN_);
    float dn = dsum[b * NN_ + n], dm = dsum[b * NN_ + m];
    float in = dn > 0.f ? rsqrtf(fmaxf(dn, 1e-12f)) : 0.f;
    float im = dm > 0.f ? rsqrtf(fmaxf(dm, 1e-12f)) : 0.f;
    float v = ((n == m) ? 1.f : 0.f) - in * A[idx] * im;
    L[idx] = (__bf16)v;
}

// ---------------------------------------------------------------------------
// Reeb segment-max pool: Vf[b,r,f] = max_s out[b, sccs[b,r,s], f]
// ---------------------------------------------------------------------------
__global__ void reeb_pool(const float* __restrict__ X, const int* __restrict__ sccs,
                          float* __restrict__ Vf)
{
    int br = blockIdx.x;                 // b*R + r
    int b  = br / RR;
    int f  = threadIdx.x;                // 128 threads
    const int* sc = sccs + (long long)br * SS;
    float m = -3.4e38f;
    for (int s = 0; s < SS; ++s) {
        int idx = sc[s];
        idx = idx < 0 ? 0 : (idx >= NN_ ? NN_ - 1 : idx);
        m = fmaxf(m, X[((long long)b * NN_ + idx) * 128 + f]);
    }
    Vf[(long long)br * 128 + f] = m;
}

// max over rows: P[b,f] = max_n X[b,n,f]
__global__ void maxpool_rows(const float* __restrict__ X, float* __restrict__ P,
                             int rows, int F)
{
    int b = blockIdx.x, f = threadIdx.x;
    if (f >= F) return;
    float m = -3.4e38f;
    const float* p = X + (long long)b * rows * F + f;
    for (int n = 0; n < rows; ++n) m = fmaxf(m, p[(long long)n * F]);
    P[(long long)b * F + f] = m;
}

__global__ void concat_pools(const float* __restrict__ pR, const float* __restrict__ pN,
                             float* __restrict__ cat)
{
    int i = blockIdx.x * blockDim.x + threadIdx.x;
    if (i >= BB * 1024) return;
    int b = i / 1024, j = i % 1024;
    cat[i] = (j < 512) ? pR[b * 512 + j] : pN[b * 512 + (j - 512)];
}

__global__ void fc_layer(const float* __restrict__ in, const float* __restrict__ W,
                         const float* __restrict__ bias, float* __restrict__ out,
                         int Bn, int K, int O, int relu)
{
    int i = blockIdx.x * blockDim.x + threadIdx.x;
    if (i >= Bn * O) return;
    int b = i / O, o = i % O;
    const float* ip = in + (long long)b * K;
    float s = bias[o];
    for (int k = 0; k < K; ++k) s += ip[k] * W[(long long)k * O + o];
    out[i] = relu ? fmaxf(s, 0.f) : s;
}

__global__ void zero_regs(float* __restrict__ out)
{
    int i = threadIdx.x;
    if (i < 9) out[96 + i] = 0.f;
}

__global__ void sumsq_atomic(const float* __restrict__ X, long long n,
                             float* __restrict__ target)
{
    __shared__ float red[256];
    float s = 0.f;
    for (long long i = (long long)blockIdx.x * 256 + threadIdx.x; i < n;
         i += (long long)gridDim.x * 256)
        s += X[i] * X[i];
    red[threadIdx.x] = s; __syncthreads();
    for (int t = 128; t; t >>= 1) {
        if (threadIdx.x < t) red[threadIdx.x] += red[threadIdx.x + t];
        __syncthreads();
    }
    if (threadIdx.x == 0) atomicAdd(target, red[0]);
}

// ||W[:,0]||^2 and b[0]^2 for the three FC layers -> out[99..104]
__global__ void weight_regs(const float* w1, const float* b1,
                            const float* w2, const float* b2,
                            const float* w3, const float* b3,
                            float* __restrict__ out)
{
    __shared__ float red[256];
    int tid = threadIdx.x;
    float s = 0.f;
    for (int i = tid; i < 1024; i += 256) { float v = w1[(long long)i * 512]; s += v * v; }
    red[tid] = s; __syncthreads();
    for (int t = 128; t; t >>= 1) { if (tid < t) red[tid] += red[tid + t]; __syncthreads(); }
    if (tid == 0) { out[99] = red[0]; out[100] = b1[0] * b1[0]; }
    __syncthreads();
    s = 0.f;
    for (int i = tid; i < 512; i += 256) { float v = w2[(long long)i * 128]; s += v * v; }
    red[tid] = s; __syncthreads();
    for (int t = 128; t; t >>= 1) { if (tid < t) red[tid] += red[tid + t]; __syncthreads(); }
    if (tid == 0) { out[101] = red[0]; out[102] = b2[0] * b2[0]; }
    __syncthreads();
    s = 0.f;
    for (int i = tid; i < 128; i += 256) { float v = w3[(long long)i * NCLS]; s += v * v; }
    red[tid] = s; __syncthreads();
    for (int t = 128; t; t >>= 1) { if (tid < t) red[tid] += red[tid + t]; __syncthreads(); }
    if (tid == 0) { out[103] = red[0]; out[104] = b3[0] * b3[0]; }
}

// ---------------------------------------------------------------------------
// host-side GEMM dispatch
// ---------------------------------------------------------------------------
static void gemm(int ta, int tb, bool abf,
                 const void* A, const float* Bm, float* C,
                 const float* Csrc, const float* bias,
                 int M, int N, int K, int lda, int ldb, int ldc, int ldcs,
                 long long sA, long long sB, long long sC, long long sCs,
                 float alpha, float beta, int relu, int batch, hipStream_t s)
{
    dim3 grid((N + TN - 1) / TN, (M + TM - 1) / TM, batch), block(128);
    if (abf)
        gemm_wmma<0, 0, true ><<<grid, block, 0, s>>>(A, Bm, C, Csrc, bias, M, N, K,
            lda, ldb, ldc, ldcs, sA, sB, sC, sCs, alpha, beta, relu);
    else if (ta == 0 && tb == 0)
        gemm_wmma<0, 0, false><<<grid, block, 0, s>>>(A, Bm, C, Csrc, bias, M, N, K,
            lda, ldb, ldc, ldcs, sA, sB, sC, sCs, alpha, beta, relu);
    else if (ta == 0 && tb == 1)
        gemm_wmma<0, 1, false><<<grid, block, 0, s>>>(A, Bm, C, Csrc, bias, M, N, K,
            lda, ldb, ldc, ldcs, sA, sB, sC, sCs, alpha, beta, relu);
    else
        gemm_wmma<1, 0, false><<<grid, block, 0, s>>>(A, Bm, C, Csrc, bias, M, N, K,
            lda, ldb, ldc, ldcs, sA, sB, sC, sCs, alpha, beta, relu);
}

extern "C" void kernel_launch(void* const* d_in, const int* in_sizes, int n_in,
                              void* d_out, int out_size, void* d_ws, size_t ws_size,
                              hipStream_t stream)
{
    const float* x      = (const float*)d_in[0];   // [16,2048,6]
    const float* lapR   = (const float*)d_in[1];   // [16,64,64]
    const int*   sccs   = (const int*)  d_in[2];   // [16,64,64]
    const float* w1     = (const float*)d_in[9];   // [18,128]
    const float* b1     = (const float*)d_in[10];
    const float* w2     = (const float*)d_in[11];  // [384,512]
    const float* b2     = (const float*)d_in[12];
    const float* wR     = (const float*)d_in[13];  // [384,512]
    const float* bR     = (const float*)d_in[14];
    const float* fw1    = (const float*)d_in[15];  // [1024,512]
    const float* fb1    = (const float*)d_in[16];
    const float* fw2    = (const float*)d_in[17];  // [512,128]
    const float* fb2    = (const float*)d_in[18];
    const float* fw3    = (const float*)d_in[19];  // [128,6]
    const float* fb3    = (const float*)d_in[20];
    float* out = (float*)d_out;

    // ---- workspace carve-up (sequentially reused across the two epochs) ----
    char* w = (char*)d_ws;
    auto carve = [&](size_t bytes) { void* p = (void*)w; w += (bytes + 255) & ~(size_t)255; return p; };
    float*  G    = (float*) carve((size_t)BB * NN_ * NN_ * 4);   // gram / sparsified A
    __bf16* Lb   = (__bf16*)carve((size_t)BB * NN_ * NN_ * 2);   // Laplacian, bf16
    float*  xx   = (float*) carve((size_t)BB * NN_ * 4);
    float*  dsum = (float*) carve((size_t)BB * NN_ * 4);
    float*  t1b  = (float*) carve((size_t)BB * NN_ * 512 * 4);   // Cheb t1
    float*  t2b  = (float*) carve((size_t)BB * NN_ * 512 * 4);   // Cheb t2
    float*  out1 = (float*) carve((size_t)BB * NN_ * 128 * 4);
    float*  out2 = (float*) carve((size_t)BB * NN_ * 512 * 4);
    float*  Y    = (float*) carve((size_t)BB * NN_ * 512 * 4);   // L*out for regs
    float*  Z    = (float*) carve((size_t)BB * 512 * 512 * 4);   // out^T*(L*out)
    float*  Vf   = (float*) carve((size_t)BB * RR * 128 * 4);
    float*  t1r  = (float*) carve((size_t)BB * RR * 128 * 4);
    float*  t2r  = (float*) carve((size_t)BB * RR * 128 * 4);
    float*  outR = (float*) carve((size_t)BB * RR * 512 * 4);
    float*  Yr   = (float*) carve((size_t)BB * RR * 512 * 4);
    float*  pN   = (float*) carve((size_t)BB * 512 * 4);
    float*  pR   = (float*) carve((size_t)BB * 512 * 4);
    float*  cat  = (float*) carve((size_t)BB * 1024 * 4);
    float*  h1   = (float*) carve((size_t)BB * 512 * 4);
    float*  h2   = (float*) carve((size_t)BB * 128 * 4);

    const long long sNN = (long long)NN_ * NN_;
    const long long total_nn = (long long)BB * NN_ * NN_;

    zero_regs<<<1, 32, 0, stream>>>(out);

    // ================= epoch 1: L1 from x (F=6) =================
    sqnorm_rows<<<(BB * NN_ + 255) / 256, 256, 0, stream>>>(x, xx, (long long)BB * NN_, F0);
    gemm(0, 1, false, x, x, G, nullptr, nullptr, NN_, NN_, F0, F0, F0, NN_, 0,
         (long long)NN_ * F0, (long long)NN_ * F0, sNN, 0, 1.f, 0.f, 0, BB, stream);
    affinity_topk<<<BB * NN_, 256, 0, stream>>>(G, xx, dsum, KNN);
    build_lap_bf16<<<(int)((total_nn + 255) / 256), 256, 0, stream>>>(G, dsum, Lb);

    // t1 = L*x ; t2 = 2*L*t1 - x   (bf16-A WMMA, N=6 padded in-tile)
    gemm(0, 0, true, Lb, x, t1b, nullptr, nullptr, NN_, F0, NN_, NN_, F0, F0, 0,
         sNN, (long long)NN_ * F0, (long long)NN_ * F0, 0, 1.f, 0.f, 0, BB, stream);
    gemm(0, 0, true, Lb, t1b, t2b, x, nullptr, NN_, F0, NN_, NN_, F0, F0, F0,
         sNN, (long long)NN_ * F0, (long long)NN_ * F0, (long long)NN_ * F0, 2.f, -1.f, 0, BB, stream);
    // conv1: out1 = relu([t0,t1,t2] W1 + b1), accumulated in 3 K=6 slices
    gemm(0, 0, false, x,   w1,             out1, nullptr, nullptr, NN_, 128, F0, F0, 128, 128, 0,
         (long long)NN_ * F0, 0, (long long)NN_ * 128, 0, 1.f, 0.f, 0, BB, stream);
    gemm(0, 0, false, t1b, w1 + 6 * 128,   out1, out1, nullptr, NN_, 128, F0, F0, 128, 128, 128,
         (long long)NN_ * F0, 0, (long long)NN_ * 128, (long long)NN_ * 128, 1.f, 1.f, 0, BB, stream);
    gemm(0, 0, false, t2b, w1 + 12 * 128,  out1, out1, b1, NN_, 128, F0, F0, 128, 128, 128,
         (long long)NN_ * F0, 0, (long long)NN_ * 128, (long long)NN_ * 128, 1.f, 1.f, 1, BB, stream);

    // reg1 = || out1^T (L1 out1) ||^2
    gemm(0, 0, true, Lb, out1, Y, nullptr, nullptr, NN_, 128, NN_, NN_, 128, 128, 0,
         sNN, (long long)NN_ * 128, (long long)NN_ * 128, 0, 1.f, 0.f, 0, BB, stream);
    gemm(1, 0, false, out1, Y, Z, nullptr, nullptr, 128, 128, NN_, 128, 128, 128, 0,
         (long long)NN_ * 128, (long long)NN_ * 128, (long long)128 * 128, 0, 1.f, 0.f, 0, BB, stream);
    sumsq_atomic<<<512, 256, 0, stream>>>(Z, (long long)BB * 128 * 128, out + 96);

    // Reeb segment-max pool
    reeb_pool<<<BB * RR, 128, 0, stream>>>(out1, sccs, Vf);

    // ================= epoch 2: L2 from out1 (F=128) =================
    sqnorm_rows<<<(BB * NN_ + 255) / 256, 256, 0, stream>>>(out1, xx, (long long)BB * NN_, 128);
    gemm(0, 1, false, out1, out1, G, nullptr, nullptr, NN_, NN_, 128, 128, 128, NN_, 0,
         (long long)NN_ * 128, (long long)NN_ * 128, sNN, 0, 1.f, 0.f, 0, BB, stream);
    affinity_topk<<<BB * NN_, 256, 0, stream>>>(G, xx, dsum, KNN);
    build_lap_bf16<<<(int)((total_nn + 255) / 256), 256, 0, stream>>>(G, dsum, Lb);

    gemm(0, 0, true, Lb, out1, t1b, nullptr, nullptr, NN_, 128, NN_, NN_, 128, 128, 0,
         sNN, (long long)NN_ * 128, (long long)NN_ * 128, 0, 1.f, 0.f, 0, BB, stream);
    gemm(0, 0, true, Lb, t1b, t2b, out1, nullptr, NN_, 128, NN_, NN_, 128, 128, 128,
         sNN, (long long)NN_ * 128, (long long)NN_ * 128, (long long)NN_ * 128, 2.f, -1.f, 0, BB, stream);
    // conv2 (K=128 slices of W2[384,512])
    gemm(0, 0, false, out1, w2,              out2, nullptr, nullptr, NN_, 512, 128, 128, 512, 512, 0,
         (long long)NN_ * 128, 0, (long long)NN_ * 512, 0, 1.f, 0.f, 0, BB, stream);
    gemm(0, 0, false, t1b,  w2 + 128 * 512,  out2, out2, nullptr, NN_, 512, 128, 128, 512, 512, 512,
         (long long)NN_ * 128, 0, (long long)NN_ * 512, (long long)NN_ * 512, 1.f, 1.f, 0, BB, stream);
    gemm(0, 0, false, t2b,  w2 + 256 * 512,  out2, out2, b2, NN_, 512, 128, 128, 512, 512, 512,
         (long long)NN_ * 128, 0, (long long)NN_ * 512, (long long)NN_ * 512, 1.f, 1.f, 1, BB, stream);

    // reg2
    gemm(0, 0, true, Lb, out2, Y, nullptr, nullptr, NN_, 512, NN_, NN_, 512, 512, 0,
         sNN, (long long)NN_ * 512, (long long)NN_ * 512, 0, 1.f, 0.f, 0, BB, stream);
    gemm(1, 0, false, out2, Y, Z, nullptr, nullptr, 512, 512, NN_, 512, 512, 512, 0,
         (long long)NN_ * 512, (long long)NN_ * 512, (long long)512 * 512, 0, 1.f, 0.f, 0, BB, stream);
    sumsq_atomic<<<512, 256, 0, stream>>>(Z, (long long)BB * 512 * 512, out + 97);

    // ================= Reeb branch (R=64, laplacian given) =================
    gemm(0, 0, false, lapR, Vf, t1r, nullptr, nullptr, RR, 128, RR, RR, 128, 128, 0,
         (long long)RR * RR, (long long)RR * 128, (long long)RR * 128, 0, 1.f, 0.f, 0, BB, stream);
    gemm(0, 0, false, lapR, t1r, t2r, Vf, nullptr, RR, 128, RR, RR, 128, 128, 128,
         (long long)RR * RR, (long long)RR * 128, (long long)RR * 128, (long long)RR * 128, 2.f, -1.f, 0, BB, stream);
    gemm(0, 0, false, Vf,  wR,             outR, nullptr, nullptr, RR, 512, 128, 128, 512, 512, 0,
         (long long)RR * 128, 0, (long long)RR * 512, 0, 1.f, 0.f, 0, BB, stream);
    gemm(0, 0, false, t1r, wR + 128 * 512, outR, outR, nullptr, RR, 512, 128, 128, 512, 512, 512,
         (long long)RR * 128, 0, (long long)RR * 512, (long long)RR * 512, 1.f, 1.f, 0, BB, stream);
    gemm(0, 0, false, t2r, wR + 256 * 512, outR, outR, bR, RR, 512, 128, 128, 512, 512, 512,
         (long long)RR * 128, 0, (long long)RR * 512, (long long)RR * 512, 1.f, 1.f, 1, BB, stream);

    // reg3
    gemm(0, 0, false, lapR, outR, Yr, nullptr, nullptr, RR, 512, RR, RR, 512, 512, 0,
         (long long)RR * RR, (long long)RR * 512, (long long)RR * 512, 0, 1.f, 0.f, 0, BB, stream);
    gemm(1, 0, false, outR, Yr, Z, nullptr, nullptr, 512, 512, RR, 512, 512, 512, 0,
         (long long)RR * 512, (long long)RR * 512, (long long)512 * 512, 0, 1.f, 0.f, 0, BB, stream);
    sumsq_atomic<<<512, 256, 0, stream>>>(Z, (long long)BB * 512 * 512, out + 98);

    // ================= head =================
    maxpool_rows<<<BB, 512, 0, stream>>>(out2, pN, NN_, 512);
    maxpool_rows<<<BB, 512, 0, stream>>>(outR, pR, RR, 512);
    concat_pools<<<(BB * 1024 + 255) / 256, 256, 0, stream>>>(pR, pN, cat);
    fc_layer<<<(BB * 512 + 255) / 256, 256, 0, stream>>>(cat, fw1, fb1, h1, BB, 1024, 512, 1);
    fc_layer<<<(BB * 128 + 255) / 256, 256, 0, stream>>>(h1, fw2, fb2, h2, BB, 512, 128, 1);
    fc_layer<<<(BB * NCLS + 255) / 256, 256, 0, stream>>>(h2, fw3, fb3, out, BB, 128, NCLS, 0);

    weight_regs<<<1, 256, 0, stream>>>(fw1, fb1, fw2, fb2, fw3, fb3, out);

    (void)in_sizes; (void)n_in; (void)out_size; (void)ws_size;
}
